// RecurrentDynamics_38173669327210
// MI455X (gfx1250) — compile-verified, matrix operand order checked by reading, not run on hardware
//
#include <hip/hip_runtime.h>

// ---------------------------------------------------------------------------
// Types for CDNA5 WMMA (wave32): v_wmma_f32_16x16x32_bf16
// ---------------------------------------------------------------------------
typedef __attribute__((ext_vector_type(16))) __bf16 v16bf;
typedef __attribute__((ext_vector_type(8)))  float  v8f;
typedef __attribute__((ext_vector_type(4)))  unsigned int u32x4;
typedef __attribute__((ext_vector_type(4)))  float  f32x4;

union Frag {
  v16bf v;
  u32x4 q[2];
  __bf16 e[16];
};

// A/B fragment (16x32 bf16): lane L<16 -> row L, K pairs {0..7},{16..23};
// lane L>=16 -> row L-16, K pairs {8..15},{24..31}. With hi = (L<16?0:8):
// two contiguous 16B loads at (k0+hi) and (k0+hi+16).
__device__ inline void frag_ld_bf16(Frag& f, const __bf16* p) {
  f.q[0] = *reinterpret_cast<const u32x4*>(p);
  f.q[1] = *reinterpret_cast<const u32x4*>(p + 16);
}
__device__ inline void frag_ld_f32(Frag& f, const float* p) {
  f32x4 a0 = *reinterpret_cast<const f32x4*>(p);
  f32x4 a1 = *reinterpret_cast<const f32x4*>(p + 4);
  f32x4 a2 = *reinterpret_cast<const f32x4*>(p + 16);
  f32x4 a3 = *reinterpret_cast<const f32x4*>(p + 20);
#pragma unroll
  for (int i = 0; i < 4; ++i) {
    f.e[i]      = (__bf16)a0[i];
    f.e[4 + i]  = (__bf16)a1[i];
    f.e[8 + i]  = (__bf16)a2[i];
    f.e[12 + i] = (__bf16)a3[i];
  }
}
__device__ inline v8f wmma_bf16(const Frag& a, const Frag& b, v8f c) {
  return __builtin_amdgcn_wmma_f32_16x16x32_bf16(
      /*neg_a=*/false, a.v, /*neg_b=*/false, b.v,
      /*c_mod=*/(short)0, c, /*reuse_a=*/false, /*reuse_b=*/false);
}

template <int ACT> __device__ inline float apply_act(float x) {
  if (ACT == 1) return x > 0.f ? x : (expf(x) - 1.f);  // elu
  if (ACT == 2) return x > 0.f ? x : 0.f;              // relu
  return x;
}
__device__ inline float sigmoidf(float x) { return 1.f / (1.f + expf(-x)); }

// CDNA5 async global->LDS copy (16B per lane, ASYNCcnt tracked)
__device__ inline void async_ld_lds_b128(unsigned lds_off, const void* gptr) {
  unsigned long long ga = (unsigned long long)(uintptr_t)gptr;
  asm volatile("global_load_async_to_lds_b128 %0, %1, off"
               :: "v"(lds_off), "v"(ga) : "memory");
}
__device__ inline void wait_asynccnt0() {
  asm volatile("s_wait_asynccnt 0x0" ::: "memory");
}

// ---------------------------------------------------------------------------
// Generic WMMA GEMM: C[M,N] = act(A[M,K] @ Wt^T + bias)   (Wt is N x K bf16)
// Block: 256 thr = 8 waves, tile 128x256; wave = 4 M-sub x 4 N-sub (64x64).
// bf16-A path: A tile (128x32) staged via GLOBAL_LOAD_ASYNC_TO_LDS_B128.
// f32-A path (raw input only): direct global loads + inline cvt to bf16.
// ---------------------------------------------------------------------------
template <int ACT, bool AF32, bool OUTBF>
__global__ __launch_bounds__(256) void gemm_wmma_kernel(
    const void* __restrict__ Ap, const __bf16* __restrict__ Wt,
    const float* __restrict__ bias, void* __restrict__ Cp,
    int M, int N, int K) {
  __shared__ __attribute__((aligned(16))) __bf16 Alds[128 * 32];  // 8 KB
  const int tid = threadIdx.x;
  const int wave = tid >> 5, l = tid & 31, l16 = l & 15;
  const int hi = (l & 16) ? 8 : 0;
  const int wm = wave & 1, wn = wave >> 1;  // 2 x 4 wave grid (64x64 each)
  const int bm = blockIdx.y * 128, bn = blockIdx.x * 256;
  const unsigned lds0 = (unsigned)(uintptr_t)(const void*)Alds;  // LDS byte off

  v8f acc[4][4];
#pragma unroll
  for (int i = 0; i < 4; ++i)
#pragma unroll
    for (int j = 0; j < 4; ++j)
#pragma unroll
      for (int e = 0; e < 8; ++e) acc[i][j][e] = 0.f;

  for (int k0 = 0; k0 < K; k0 += 32) {
    if (!AF32) {
      // Stage A tile 128x32 bf16 = 8KB: 512 16B chunks, 2 per thread.
#pragma unroll
      for (int c = tid; c < 512; c += 256) {
        int row = bm + (c >> 2);
        if (row >= M) row = M - 1;
        const __bf16* gp = (const __bf16*)Ap + (size_t)row * K + k0 + (c & 3) * 8;
        async_ld_lds_b128(lds0 + (unsigned)c * 16u, gp);
      }
      wait_asynccnt0();
      __syncthreads();
    }

    Frag a[4], b[4];
#pragma unroll
    for (int i = 0; i < 4; ++i) {
      if (AF32) {
        int r = bm + (wm * 64 + i * 16) + l16;
        if (r >= M) r = M - 1;
        frag_ld_f32(a[i], (const float*)Ap + (size_t)r * K + k0 + hi);
      } else {
        int lr = wm * 64 + i * 16 + l16;  // local row in staged tile
        frag_ld_bf16(a[i], &Alds[lr * 32 + hi]);
      }
    }
#pragma unroll
    for (int j = 0; j < 4; ++j) {
      int n = bn + wn * 64 + j * 16 + l16;
      if (n >= N) n = N - 1;
      const __bf16* wp = Wt + (size_t)n * K + k0 + hi;
      frag_ld_bf16(b[j], wp);
      if (k0 + 32 < K) __builtin_prefetch(wp + 32, 0, 1);  // global_prefetch_b8
    }
#pragma unroll
    for (int i = 0; i < 4; ++i)
#pragma unroll
      for (int j = 0; j < 4; ++j) acc[i][j] = wmma_bf16(a[i], b[j], acc[i][j]);

    if (!AF32) __syncthreads();  // protect Alds before next stage
  }

  // C layout: lane l -> col l&15; VGPR p -> row p + (l<16?0:8) (== hi)
#pragma unroll
  for (int i = 0; i < 4; ++i)
#pragma unroll
    for (int j = 0; j < 4; ++j) {
      int n = bn + wn * 64 + j * 16 + l16;
      if (n >= N) continue;
      float bv = bias[n];
#pragma unroll
      for (int p = 0; p < 8; ++p) {
        int r = bm + wm * 64 + i * 16 + p + hi;
        if (r >= M) continue;
        float v = apply_act<ACT>(acc[i][j][p] + bv);
        if (OUTBF)
          ((__bf16*)Cp)[(size_t)r * N + n] = (__bf16)v;
        else
          ((float*)Cp)[(size_t)r * N + n] = v;
      }
    }
}

// ---------------------------------------------------------------------------
// Weight transpose + f32 -> bf16:  W (K x N, row major) -> Wt (N x K bf16)
// ---------------------------------------------------------------------------
__global__ void transpose_bf16_kernel(const float* __restrict__ W,
                                      __bf16* __restrict__ Wt, int K, int N) {
  int total = K * N;
  for (int idx = blockIdx.x * blockDim.x + threadIdx.x; idx < total;
       idx += gridDim.x * blockDim.x) {
    int n = idx / K, k = idx - n * K;
    Wt[idx] = (__bf16)W[(size_t)k * N + n];
  }
}

// Gather time-reversed prefix rows: out[b*Tsub+t] = s[b*256 + (Tsub-1-t)]
__global__ void gather_rev_kernel(const __bf16* __restrict__ s,
                                  __bf16* __restrict__ out, int Tsub) {
  int total = 128 * Tsub * 256;
  for (int idx = blockIdx.x * blockDim.x + threadIdx.x; idx < total;
       idx += gridDim.x * blockDim.x) {
    int j = idx & 255;
    int r = idx >> 8;
    int b = r / Tsub, tt = r - b * Tsub;
    out[idx] = s[((size_t)b * 256 + (Tsub - 1 - tt)) * 256 + j];
  }
}

// ---------------------------------------------------------------------------
// GRU: block owns 16 batch rows; gh = h @ Wh + bh via WMMA each step.
// Only the time-mean of h is consumed downstream -> accumulate in registers.
// ---------------------------------------------------------------------------
__global__ __launch_bounds__(256) void gru_wmma_kernel(
    const float* __restrict__ xp,      // (128*T) x 768, rows = b*T + t
    const __bf16* __restrict__ Wht,    // 768 x 256 (transposed bf16)
    const float* __restrict__ bh,      // 768
    float* __restrict__ hbar,          // 128 x 256
    int T) {
  __shared__ __attribute__((aligned(16))) __bf16 hb[16 * 256];
  __shared__ __attribute__((aligned(16))) __bf16 ghs[16 * 768];
  const int tid = threadIdx.x;
  const int wave = tid >> 5, l = tid & 31, l16 = l & 15;
  const int hi = (l & 16) ? 8 : 0;
  const int br0 = blockIdx.x * 16;

  float h[16], hs[16];
#pragma unroll
  for (int b = 0; b < 16; ++b) { h[b] = 0.f; hs[b] = 0.f; }
  for (int idx = tid; idx < 16 * 256; idx += 256) hb[idx] = (__bf16)0.f;
  __syncthreads();

  for (int t = 0; t < T; ++t) {
    // gh = h @ Wh + bh  : M=16 (1 subtile), N=768 (48 subtiles / 8 waves)
    for (int s = 0; s < 6; ++s) {
      int nsub = wave * 6 + s;
      int n = nsub * 16 + l16;
      v8f acc;
#pragma unroll
      for (int e = 0; e < 8; ++e) acc[e] = 0.f;
      for (int k0 = 0; k0 < 256; k0 += 32) {
        Frag a, b;
        frag_ld_bf16(a, &hb[l16 * 256 + k0 + hi]);
        frag_ld_bf16(b, Wht + (size_t)n * 256 + k0 + hi);
        acc = wmma_bf16(a, b, acc);
      }
      float bv = bh[n];
#pragma unroll
      for (int p = 0; p < 8; ++p) ghs[(p + hi) * 768 + n] = (__bf16)(acc[p] + bv);
    }
    __syncthreads();

    // gating: thread owns hidden column j = tid for all 16 batch rows
    int j = tid;
#pragma unroll
    for (int bb = 0; bb < 16; ++bb) {
      size_t xrow = ((size_t)(br0 + bb) * T + t) * 768;
      float xr = xp[xrow + j], xz = xp[xrow + 256 + j], xn = xp[xrow + 512 + j];
      float hr = (float)ghs[bb * 768 + j];
      float hz = (float)ghs[bb * 768 + 256 + j];
      float hn = (float)ghs[bb * 768 + 512 + j];
      float r = sigmoidf(xr + hr);
      float z = sigmoidf(xz + hz);
      float n2 = tanhf(xn + r * hn);
      float hn2 = (1.f - z) * n2 + z * h[bb];
      h[bb] = hn2;
      hs[bb] += hn2;
      hb[bb * 256 + j] = (__bf16)hn2;
    }
    __syncthreads();
  }
#pragma unroll
  for (int bb = 0; bb < 16; ++bb)
    hbar[(size_t)(br0 + bb) * 256 + tid] = hs[bb] / (float)T;
}

// ---------------------------------------------------------------------------
// Head kernels (tiny GEMVs, VALU is fine)
// ---------------------------------------------------------------------------
__global__ void agg_head_kernel(const float* __restrict__ hbar,
                                const float* __restrict__ aggW, const float* __restrict__ aggb,
                                const float* __restrict__ muW, const float* __restrict__ mub,
                                const float* __restrict__ varW, const float* __restrict__ varb,
                                float* __restrict__ mu_c, float* __restrict__ var_c,
                                float* __restrict__ zc) {
  __shared__ float zagg[128 * 64];
  for (int idx = threadIdx.x; idx < 8192; idx += blockDim.x) {
    int b = idx >> 6, j = idx & 63;
    float acc = aggb[j];
    for (int k = 0; k < 256; ++k) acc += hbar[b * 256 + k] * aggW[k * 64 + j];
    zagg[idx] = acc;
  }
  __syncthreads();
  for (int idx = threadIdx.x; idx < 8192; idx += blockDim.x) {
    int b = idx >> 6, j = idx & 63;
    float m = mub[j], v = varb[j];
    for (int k = 0; k < 64; ++k) {
      float z = zagg[b * 64 + k];
      m += z * muW[k * 64 + j];
      v += z * varW[k * 64 + j];
    }
    m = fminf(fmaxf(m, -100.f), 85.f);
    v = tanhf(fminf(fmaxf(v, -100.f), 85.f));
    mu_c[idx] = m;
    var_c[idx] = v;
    zc[idx] = m;
  }
}

__global__ void init_head_kernel(const float* __restrict__ hbar,
                                 const float* __restrict__ imuW, const float* __restrict__ imub,
                                 const float* __restrict__ ivarW, const float* __restrict__ ivarb,
                                 float* __restrict__ mu0, float* __restrict__ var0,
                                 float* __restrict__ z0) {
  for (int idx = threadIdx.x; idx < 8192; idx += blockDim.x) {
    int b = idx >> 6, j = idx & 63;
    float m = imub[j], v = ivarb[j];
    for (int k = 0; k < 256; ++k) {
      float hh = hbar[b * 256 + k];
      m += hh * imuW[k * 64 + j];
      v += hh * ivarW[k * 64 + j];
    }
    mu0[idx] = m;
    z0[idx] = m;
    var0[idx] = tanhf(v);
  }
}

// ---------------------------------------------------------------------------
// Latent transition scan: 255 sequential steps, WMMA for t1/t2/t3/t4/tz.
// Block owns 16 batch rows (batch rows are independent).
// ---------------------------------------------------------------------------
__global__ __launch_bounds__(256) void scan_wmma_kernel(
    const float* __restrict__ z0, const float* __restrict__ zc,
    const __bf16* __restrict__ t1t, const float* __restrict__ t1b,
    const __bf16* __restrict__ t2t, const float* __restrict__ t2b,
    const __bf16* __restrict__ t3t, const float* __restrict__ t3b,
    const __bf16* __restrict__ t4t, const float* __restrict__ t4b,
    const __bf16* __restrict__ tzt, const float* __restrict__ tzb,
    __bf16* __restrict__ zseq) {
  __shared__ __attribute__((aligned(16))) __bf16 zz[16 * 128];   // [z | zc]
  __shared__ __attribute__((aligned(16))) __bf16 ht1[16 * 256];
  __shared__ __attribute__((aligned(16))) __bf16 ht3[16 * 256];
  __shared__ float gate[16 * 64];
  __shared__ float prop[16 * 64];
  __shared__ float tzo[16 * 64];
  const int tid = threadIdx.x;
  const int wave = tid >> 5, l = tid & 31, l16 = l & 15;
  const int hi = (l & 16) ? 8 : 0;
  const int br0 = blockIdx.x * 16;

  for (int idx = tid; idx < 16 * 64; idx += 256) {
    int b = idx >> 6, j = idx & 63;
    float zv = z0[(size_t)(br0 + b) * 64 + j];
    float cv = zc[(size_t)(br0 + b) * 64 + j];
    zz[b * 128 + j] = (__bf16)zv;
    zz[b * 128 + 64 + j] = (__bf16)cv;
    zseq[((size_t)(br0 + b) * 256) * 64 + j] = (__bf16)zv;
  }
  __syncthreads();

  for (int t = 1; t < 256; ++t) {
    // phase 1: t1 (16 subtiles), t3 (16), tz (4) -> 36 subtiles over 8 waves
    for (int id = wave; id < 36; id += 8) {
      v8f acc;
#pragma unroll
      for (int e = 0; e < 8; ++e) acc[e] = 0.f;
      if (id < 32) {
        const __bf16* Wp = (id < 16) ? t1t : t3t;
        int n = (id & 15) * 16 + l16;
        for (int k0 = 0; k0 < 128; k0 += 32) {
          Frag a, b;
          frag_ld_bf16(a, &zz[l16 * 128 + k0 + hi]);
          frag_ld_bf16(b, Wp + (size_t)n * 128 + k0 + hi);
          acc = wmma_bf16(a, b, acc);
        }
        float bv = (id < 16) ? t1b[n] : t3b[n];
        __bf16* dst = (id < 16) ? ht1 : ht3;
#pragma unroll
        for (int p = 0; p < 8; ++p) {
          float v = acc[p] + bv;
          dst[(p + hi) * 256 + n] = (__bf16)(v > 0.f ? v : 0.f);  // relu
        }
      } else {
        int n = (id - 32) * 16 + l16;
        for (int k0 = 0; k0 < 64; k0 += 32) {
          Frag a, b;
          frag_ld_bf16(a, &zz[l16 * 128 + k0 + hi]);  // z part only (K=64)
          frag_ld_bf16(b, tzt + (size_t)n * 64 + k0 + hi);
          acc = wmma_bf16(a, b, acc);
        }
        float bv = tzb[n];
#pragma unroll
        for (int p = 0; p < 8; ++p) tzo[(p + hi) * 64 + n] = acc[p] + bv;
      }
    }
    __syncthreads();

    // phase 2: t2 (gate, waves 0-3) and t4 (prop, waves 4-7), K=256
    {
      const __bf16* Ap = (wave < 4) ? ht1 : ht3;
      const __bf16* Wp = (wave < 4) ? t2t : t4t;
      const float* bb = (wave < 4) ? t2b : t4b;
      int n = (wave & 3) * 16 + l16;
      v8f acc;
#pragma unroll
      for (int e = 0; e < 8; ++e) acc[e] = 0.f;
      for (int k0 = 0; k0 < 256; k0 += 32) {
        Frag a, b;
        frag_ld_bf16(a, Ap + l16 * 256 + k0 + hi);
        frag_ld_bf16(b, Wp + (size_t)n * 256 + k0 + hi);
        acc = wmma_bf16(a, b, acc);
      }
      float bv = bb[n];
      if (wave < 4) {
#pragma unroll
        for (int p = 0; p < 8; ++p) gate[(p + hi) * 64 + n] = sigmoidf(acc[p] + bv);
      } else {
#pragma unroll
        for (int p = 0; p < 8; ++p) prop[(p + hi) * 64 + n] = acc[p] + bv;
      }
    }
    __syncthreads();

    // phase 3: z_new = (1-g)*tz(z) + g*prop
    for (int idx = tid; idx < 16 * 64; idx += 256) {
      int b = idx >> 6, j = idx & 63;
      float g = gate[idx];
      float zn = (1.f - g) * tzo[idx] + g * prop[idx];
      zz[b * 128 + j] = (__bf16)zn;
      zseq[((size_t)(br0 + b) * 256 + t) * 64 + j] = (__bf16)zn;
    }
    __syncthreads();
  }
}

// ---------------------------------------------------------------------------
// Host launcher
// ---------------------------------------------------------------------------
extern "C" void kernel_launch(void* const* d_in, const int* in_sizes, int n_in,
                              void* d_out, int out_size, void* d_ws, size_t ws_size,
                              hipStream_t stream) {
  (void)in_sizes; (void)n_in; (void)out_size; (void)ws_size;
  const float* x    = (const float*)d_in[0];
  const float* e1W  = (const float*)d_in[1];  const float* e1b  = (const float*)d_in[2];
  const float* e2W  = (const float*)d_in[3];  const float* e2b  = (const float*)d_in[4];
  const float* e3W  = (const float*)d_in[5];  const float* e3b  = (const float*)d_in[6];
  const float* rWx  = (const float*)d_in[7];  const float* rWh  = (const float*)d_in[8];
  const float* rbx  = (const float*)d_in[9];  const float* rbh  = (const float*)d_in[10];
  const float* iWx  = (const float*)d_in[11]; const float* iWh  = (const float*)d_in[12];
  const float* ibx  = (const float*)d_in[13]; const float* ibh  = (const float*)d_in[14];
  const float* aggW = (const float*)d_in[15]; const float* aggb = (const float*)d_in[16];
  const float* muW  = (const float*)d_in[17]; const float* mub  = (const float*)d_in[18];
  const float* varW = (const float*)d_in[19]; const float* varb = (const float*)d_in[20];
  const float* imuW = (const float*)d_in[21]; const float* imub = (const float*)d_in[22];
  const float* ivW  = (const float*)d_in[23]; const float* ivb  = (const float*)d_in[24];
  const float* t1W  = (const float*)d_in[25]; const float* t1b  = (const float*)d_in[26];
  const float* t2W  = (const float*)d_in[27]; const float* t2b  = (const float*)d_in[28];
  const float* t3W  = (const float*)d_in[29]; const float* t3b  = (const float*)d_in[30];
  const float* t4W  = (const float*)d_in[31]; const float* t4b  = (const float*)d_in[32];
  const float* tzW  = (const float*)d_in[33]; const float* tzb  = (const float*)d_in[34];
  const float* m1W  = (const float*)d_in[35]; const float* m1b  = (const float*)d_in[36];
  const float* m2W  = (const float*)d_in[37]; const float* m2b  = (const float*)d_in[38];
  const float* m3W  = (const float*)d_in[39]; const float* m3b  = (const float*)d_in[40];

  // workspace carving
  uintptr_t base = (uintptr_t)d_ws;
  auto take = [&](size_t bytes) -> void* {
    uintptr_t p = base;
    base += (bytes + 255) & ~(size_t)255;
    return (void*)p;
  };
  __bf16* e1t = (__bf16*)take((size_t)2048 * 1024 * 2);
  __bf16* e2t = (__bf16*)take((size_t)2048 * 2048 * 2);
  __bf16* e3t = (__bf16*)take((size_t)256 * 2048 * 2);
  __bf16* rxt = (__bf16*)take((size_t)768 * 256 * 2);
  __bf16* rht = (__bf16*)take((size_t)768 * 256 * 2);
  __bf16* ixt = (__bf16*)take((size_t)768 * 256 * 2);
  __bf16* iht = (__bf16*)take((size_t)768 * 256 * 2);
  __bf16* t1t = (__bf16*)take((size_t)256 * 128 * 2);
  __bf16* t2t = (__bf16*)take((size_t)64 * 256 * 2);
  __bf16* t3t = (__bf16*)take((size_t)256 * 128 * 2);
  __bf16* t4t = (__bf16*)take((size_t)64 * 256 * 2);
  __bf16* tzt = (__bf16*)take((size_t)64 * 64 * 2);
  __bf16* m1t = (__bf16*)take((size_t)256 * 64 * 2);
  __bf16* m2t = (__bf16*)take((size_t)256 * 256 * 2);
  __bf16* m3t = (__bf16*)take((size_t)1024 * 256 * 2);
  __bf16* h1    = (__bf16*)take((size_t)32768 * 2048 * 2);  // reused as d1
  __bf16* h2    = (__bf16*)take((size_t)32768 * 2048 * 2);  // reused as d2
  __bf16* sbf   = (__bf16*)take((size_t)32768 * 256 * 2);
  __bf16* srevd = (__bf16*)take((size_t)8192 * 256 * 2);
  __bf16* srevi = (__bf16*)take((size_t)2048 * 256 * 2);
  float* xpd   = (float*)take((size_t)8192 * 768 * 4);
  float* xpi   = (float*)take((size_t)2048 * 768 * 4);
  float* hbard = (float*)take((size_t)128 * 256 * 4);
  float* hbari = (float*)take((size_t)128 * 256 * 4);
  float* zcw   = (float*)take((size_t)8192 * 4);
  float* z0w   = (float*)take((size_t)8192 * 4);
  __bf16* zseq = (__bf16*)take((size_t)32768 * 64 * 2);

  const size_t MU0 = 33554432, VAR0 = MU0 + 8192, MUC = MU0 + 16384, VARC = MU0 + 24576;
  float* out = (float*)d_out;

  auto tr = [&](const float* W, __bf16* Wt, int K, int N) {
    int total = K * N;
    int blocks = (total + 255) / 256;
    if (blocks > 2048) blocks = 2048;
    transpose_bf16_kernel<<<blocks, 256, 0, stream>>>(W, Wt, K, N);
  };
  tr(e1W, e1t, 1024, 2048);
  tr(e2W, e2t, 2048, 2048);
  tr(e3W, e3t, 2048, 256);
  tr(rWx, rxt, 256, 768);
  tr(rWh, rht, 256, 768);
  tr(iWx, ixt, 256, 768);
  tr(iWh, iht, 256, 768);
  tr(t1W, t1t, 128, 256);
  tr(t2W, t2t, 256, 64);
  tr(t3W, t3t, 128, 256);
  tr(t4W, t4t, 256, 64);
  tr(tzW, tzt, 64, 64);
  tr(m1W, m1t, 64, 256);
  tr(m2W, m2t, 256, 256);
  tr(m3W, m3t, 256, 1024);

  // Encoder: elu chains (bf16 WMMA, f32 accumulate); block tile 128x256
  gemm_wmma_kernel<1, true, true><<<dim3(8, 256), 256, 0, stream>>>(
      x, e1t, e1b, h1, 32768, 2048, 1024);
  gemm_wmma_kernel<1, false, true><<<dim3(8, 256), 256, 0, stream>>>(
      h1, e2t, e2b, h2, 32768, 2048, 2048);
  gemm_wmma_kernel<1, false, true><<<dim3(1, 256), 256, 0, stream>>>(
      h2, e3t, e3b, sbf, 32768, 256, 2048);

  // Time-reversed prefixes for the two GRUs
  gather_rev_kernel<<<2048, 256, 0, stream>>>(sbf, srevd, 64);
  gather_rev_kernel<<<512, 256, 0, stream>>>(sbf, srevi, 16);

  // GRU input projections: xp = s_rev @ Wx + bx (f32 out)
  gemm_wmma_kernel<0, false, false><<<dim3(3, 64), 256, 0, stream>>>(
      srevd, rxt, rbx, xpd, 8192, 768, 256);
  gemm_wmma_kernel<0, false, false><<<dim3(3, 16), 256, 0, stream>>>(
      srevi, ixt, ibx, xpi, 2048, 768, 256);

  // GRU recurrences (means only)
  gru_wmma_kernel<<<8, 256, 0, stream>>>(xpd, rht, rbh, hbard, 64);
  gru_wmma_kernel<<<8, 256, 0, stream>>>(xpi, iht, ibh, hbari, 16);

  // Heads
  agg_head_kernel<<<1, 256, 0, stream>>>(hbard, aggW, aggb, muW, mub, varW, varb,
                                         out + MUC, out + VARC, zcw);
  init_head_kernel<<<1, 256, 0, stream>>>(hbari, imuW, imub, ivW, ivb,
                                          out + MU0, out + VAR0, z0w);

  // 255-step latent transition scan
  scan_wmma_kernel<<<8, 256, 0, stream>>>(z0w, zcw, t1t, t1b, t2t, t2b, t3t, t3b,
                                          t4t, t4b, tzt, tzb, zseq);

  // Decoder: z_seq -> x_rec (final GEMM writes f32 directly into d_out)
  gemm_wmma_kernel<1, false, true><<<dim3(1, 256), 256, 0, stream>>>(
      zseq, m1t, m1b, h1, 32768, 256, 64);
  gemm_wmma_kernel<1, false, true><<<dim3(1, 256), 256, 0, stream>>>(
      h1, m2t, m2b, h2, 32768, 256, 256);
  gemm_wmma_kernel<0, false, false><<<dim3(4, 256), 256, 0, stream>>>(
      h2, m3t, m3b, out, 32768, 1024, 256);
}
